// GraphAttentionLayer_64355789964002
// MI455X (gfx1250) — matrix operand order repaired and checked by reading, not compiled, not run on hardware
//
#include <hip/hip_runtime.h>
#include <hip/hip_bf16.h>
#include <stdint.h>

#define BATCH 4
#define NN    2048
#define FIN   128
#define FOUT  128
#define NH    8
#define HD    16

typedef __attribute__((ext_vector_type(16))) __bf16 v16bf;
typedef __attribute__((ext_vector_type(8)))  float  v8f;

union Frag16 {
  v16bf v;
  unsigned short u[16];
  uint32_t d[8];
  uint4 q[2];
};

// Native f32 -> bf16 convert (lets the backend use v_cvt_*bf16* hardware
// instead of bfe/add3 bit-twiddling sequences).
__device__ __forceinline__ unsigned short f2bf(float f) {
  union { __bf16 b; unsigned short s; } c;
  c.b = (__bf16)f;
  return c.s;
}

// ---------------------------------------------------------------------------
// Kernel 1: Wx = x @ W  (bf16 WMMA, f32 accum). Stores Wx^T as bf16 [B][F][N]
// and computes e_i = Wx . a_l, e_j = Wx . a_r per head via lane reductions.
// Block: 256 thr (8 waves), each wave owns a 16-row tile; 128 rows / block.
// ---------------------------------------------------------------------------
__global__ void __launch_bounds__(256) k_gat_proj(
    const float* __restrict__ x, const float* __restrict__ W,
    const float* __restrict__ a, unsigned short* __restrict__ wxT,
    float* __restrict__ ei, float* __restrict__ ej) {
  __shared__ unsigned short WtS[FIN * FOUT];  // W^T[n][k] in bf16, 32 KB

  const int t = threadIdx.x;
  for (int idx = t; idx < FIN * FOUT; idx += 256) {
    const int k = idx >> 7, n = idx & 127;
    WtS[n * FIN + k] = f2bf(W[idx]);  // coalesced read, transposed LDS write
  }
  __syncthreads();

  const int wave = t >> 5, L = t & 31, lh = L & 15, half = L >> 4;
  const int g0 = blockIdx.x * 128 + wave * 16;   // global row of tile
  const int b  = g0 >> 11;                        // N == 2048
  const int n0 = g0 & (NN - 1);
  const float* xr = x + (size_t)(g0 + lh) * FIN;  // A-matrix row = lane%16

  // A fragments for 4 K-steps of 32 (contiguous k = 16*half + s per lane)
  Frag16 af[4];
#pragma unroll
  for (int ks = 0; ks < 4; ++ks) {
    const int lo = ks * 32 + half * 16;
    const int hi = lo + 8;
    float4 p0 = *(const float4*)(xr + lo);
    float4 p1 = *(const float4*)(xr + lo + 4);
    float4 q0 = *(const float4*)(xr + hi);
    float4 q1 = *(const float4*)(xr + hi + 4);
    af[ks].u[0]  = f2bf(p0.x); af[ks].u[1]  = f2bf(p0.y);
    af[ks].u[2]  = f2bf(p0.z); af[ks].u[3]  = f2bf(p0.w);
    af[ks].u[4]  = f2bf(p1.x); af[ks].u[5]  = f2bf(p1.y);
    af[ks].u[6]  = f2bf(p1.z); af[ks].u[7]  = f2bf(p1.w);
    af[ks].u[8]  = f2bf(q0.x); af[ks].u[9]  = f2bf(q0.y);
    af[ks].u[10] = f2bf(q0.z); af[ks].u[11] = f2bf(q0.w);
    af[ks].u[12] = f2bf(q1.x); af[ks].u[13] = f2bf(q1.y);
    af[ks].u[14] = f2bf(q1.z); af[ks].u[15] = f2bf(q1.w);
  }

#pragma unroll
  for (int ct = 0; ct < 8; ++ct) {  // 8 column tiles == 8 heads (HD == 16)
    v8f acc = {};
#pragma unroll
    for (int ks = 0; ks < 4; ++ks) {
      v16bf bf = *(const v16bf*)&WtS[(ct * 16 + lh) * FIN + ks * 32 + half * 16];
      acc = __builtin_amdgcn_wmma_f32_16x16x32_bf16(
          false, af[ks].v, false, bf, (short)0, acc, false, false);
    }

    // Store Wx^T tile as bf16: lane holds col d=lh, rows v+half*8 (contiguous n)
    {
      unsigned short pk[8];
#pragma unroll
      for (int v = 0; v < 8; ++v) pk[v] = f2bf(acc[v]);
      uint4 st;
      st.x = (uint32_t)pk[0] | ((uint32_t)pk[1] << 16);
      st.y = (uint32_t)pk[2] | ((uint32_t)pk[3] << 16);
      st.z = (uint32_t)pk[4] | ((uint32_t)pk[5] << 16);
      st.w = (uint32_t)pk[6] | ((uint32_t)pk[7] << 16);
      const size_t wb = ((size_t)(b * FOUT + ct * 16 + lh)) * NN + n0 + half * 8;
      *(uint4*)&wxT[wb] = st;
    }

    // e_i / e_j: reduce acc[m,d]*a[h,d] over d (16 lanes per row-half)
    {
      const float al = a[ct * 2 * HD + lh];        // a[:, :HD]
      const float ar = a[ct * 2 * HD + HD + lh];   // a[:, HD:]
      float sI[8], sJ[8];
#pragma unroll
      for (int v = 0; v < 8; ++v) {
        float pl = acc[v] * al, pr = acc[v] * ar;
#pragma unroll
        for (int m = 1; m < 16; m <<= 1) {
          pl += __shfl_xor(pl, m, 16);
          pr += __shfl_xor(pr, m, 16);
        }
        sI[v] = pl; sJ[v] = pr;
      }
      if (lh == 0) {  // lanes 0 (rows 0-7) and 16 (rows 8-15)
        const size_t eb = ((size_t)(b * NH + ct)) * NN + n0 + half * 8;
#pragma unroll
        for (int v = 0; v < 8; ++v) { ei[eb + v] = sI[v]; ej[eb + v] = sJ[v]; }
      }
    }
  }
}

// ---------------------------------------------------------------------------
// Kernel 2: fused masked-softmax attention + aggregation (flash-style).
// Block = (batch, 16-row tile); 8 waves = 8 heads. adj tiles double-buffered
// in LDS (one barrier per chunk), e_j band staged once (64 KB LDS).
// Pass A: scalar online max/sum. Pass B: P(bf16) x Wx(bf16) WMMA accum.
// ---------------------------------------------------------------------------
__global__ void __launch_bounds__(256) k_gat_attn(
    const int* __restrict__ adj, const unsigned short* __restrict__ wxT,
    const float* __restrict__ ei, const float* __restrict__ ej,
    unsigned short* __restrict__ hbuf) {
  __shared__ float ejS[NH * NN];        // 64 KB: e_j for all heads, whole N
  __shared__ float eiS[NH * 16];
  __shared__ float redM[NH * 16 * 16];  // 8 KB  [h][r][slice]
  __shared__ float redL[NH * 16 * 16];  // 8 KB
  __shared__ float mS[NH * 16];
  __shared__ float lS[NH * 16];
  __shared__ int   adjS[2][16 * 32];    // 4 KB double-buffered adj tile

  const int t  = threadIdx.x;
  const int b  = blockIdx.x >> 7;            // 128 row-tiles per batch
  const int i0 = (blockIdx.x & 127) * 16;
  const int* adjb = adj + (size_t)b * NN * NN;

  {  // stage e_j band + e_i tile
    const float* src = ej + (size_t)b * NH * NN;
    for (int idx = t; idx < NH * NN; idx += 256) ejS[idx] = src[idx];
  }
  if (t < 128) {
    const int h = t >> 4, r = t & 15;
    eiS[t] = ei[((size_t)(b * NH + h)) * NN + i0 + r];
  }
  __syncthreads();

  // ---- Pass A: online row max & sum(exp) over adj-masked columns ----
  {
    const int r = t & 15, slice = t >> 4;  // 16 slices of 128 j's each
    const int* arow = adjb + (size_t)(i0 + r) * NN + slice * 128;
    float eir[NH], m8[NH], l8[NH];
#pragma unroll
    for (int h = 0; h < NH; ++h) { eir[h] = eiS[h * 16 + r]; m8[h] = -INFINITY; l8[h] = 0.f; }
    for (int jj = 0; jj < 128; jj += 4) {
      const int4 av4 = *(const int4*)(arow + jj);
      const int avs[4] = {av4.x, av4.y, av4.z, av4.w};
#pragma unroll
      for (int u = 0; u < 4; ++u) {
        if (avs[u]) {
          const int j = slice * 128 + jj + u;
#pragma unroll
          for (int h = 0; h < NH; ++h) {
            const float q = eir[h] + ejS[h * NN + j];
            const float s = fmaxf(q, 0.2f * q);  // leaky_relu(0.2)
            if (s > m8[h]) { l8[h] = l8[h] * __expf(m8[h] - s) + 1.f; m8[h] = s; }
            else           { l8[h] += __expf(s - m8[h]); }
          }
        }
      }
    }
#pragma unroll
    for (int h = 0; h < NH; ++h) {
      redM[(h * 16 + r) * 16 + slice] = m8[h];
      redL[(h * 16 + r) * 16 + slice] = l8[h];
    }
  }
  __syncthreads();
  if (t < 128) {  // merge 16 slices per (h, r)
    const int h = t >> 4, r = t & 15;
    float m = -INFINITY;
#pragma unroll
    for (int s = 0; s < 16; ++s) m = fmaxf(m, redM[(h * 16 + r) * 16 + s]);
    float l = 0.f;
#pragma unroll
    for (int s = 0; s < 16; ++s) {
      const float ls = redL[(h * 16 + r) * 16 + s];
      if (ls > 0.f) l += ls * __expf(redM[(h * 16 + r) * 16 + s] - m);
    }
    mS[h * 16 + r] = m;
    lS[h * 16 + r] = l;  // l == 0 for edge-less rows -> output 0 (nan_to_num)
  }
  __syncthreads();

  // ---- Pass B: acc += exp(s - m_row) * Wx   (one wave per head) ----
  const int L = t & 31, lh = L & 15, half = L >> 4, h = t >> 5;
  const float eirow = eiS[h * 16 + lh];   // A-matrix row m = lane%16
  const float mrow  = mS[h * 16 + lh];
  const unsigned short* wrow =
      wxT + ((size_t)(b * FOUT + h * HD + lh)) * NN + half * 16;  // B: col=lh

  v8f acc = {};
  const int ar2 = t >> 4;          // adj staging: row
  const int ac2 = (t * 2) & 31;    // adj staging: col pair

  // cooperative 16x32 adj tile stage (coalesced), double-buffered
  auto stage = [&](int jt) {
    const size_t off = (size_t)(i0 + ar2) * NN + jt * 32 + ac2;
    const uint2 av = *(const uint2*)(adjb + off);
    adjS[jt & 1][ar2 * 32 + ac2]     = (int)av.x;
    adjS[jt & 1][ar2 * 32 + ac2 + 1] = (int)av.y;
    if (jt + 1 < 64) __builtin_prefetch(adjb + off + 32, 0, 1);  // global_prefetch
  };

  stage(0);
  for (int jt = 0; jt < 64; ++jt) {  // 64 chunks of 32 columns
    __syncthreads();                 // chunk jt staged; start staging jt+1
    if (jt + 1 < 64) stage(jt + 1);
    const int* adjc = adjS[jt & 1];

    Frag16 pf;  // P tile 16x32 bf16 in A layout
#pragma unroll
    for (int e = 0; e < 16; ++e) {
      const int ke = half * 16 + e;
      const int j  = jt * 32 + ke;
      const int av = adjc[lh * 32 + ke];
      const float q = eirow + ejS[h * NN + j];
      const float s = fmaxf(q, 0.2f * q);
      const float p = av ? __expf(s - mrow) : 0.f;
      pf.u[e] = f2bf(p);
    }
    Frag16 bfm;  // Wx^T B fragment: 16 contiguous K per lane
    bfm.q[0] = *(const uint4*)(wrow + jt * 32);
    bfm.q[1] = *(const uint4*)(wrow + jt * 32 + 8);

    acc = __builtin_amdgcn_wmma_f32_16x16x32_bf16(
        false, pf.v, false, bfm.v, (short)0, acc, false, false);
  }

  // normalize by row sum, write h (bf16, row-major) for the output GEMM
#pragma unroll
  for (int v = 0; v < 8; ++v) {
    const int rr = v + half * 8;
    const float l  = lS[h * 16 + rr];
    const float sc = (l > 0.f) ? 1.0f / l : 0.f;
    hbuf[((size_t)(b * NN + i0 + rr)) * FOUT + h * HD + lh] = f2bf(acc[v] * sc);
  }
}

// ---------------------------------------------------------------------------
// Kernel 3: out = elu(h @ Wo^T + bo)   (bf16 WMMA, f32 accum + epilogue)
// B[k,n] = Wo[n,k]: row-major Wo already gives contiguous-K B fragments.
// ---------------------------------------------------------------------------
__global__ void __launch_bounds__(256) k_gat_out(
    const unsigned short* __restrict__ hbuf, const float* __restrict__ Wo,
    const float* __restrict__ bo, float* __restrict__ out) {
  __shared__ unsigned short WoS[FOUT * FIN];  // 32 KB

  const int t = threadIdx.x;
  for (int idx = t; idx < FOUT * FIN; idx += 256) WoS[idx] = f2bf(Wo[idx]);
  __syncthreads();

  const int wave = t >> 5, L = t & 31, lh = L & 15, half = L >> 4;
  const int g0 = blockIdx.x * 128 + wave * 16;
  const unsigned short* hr = hbuf + (size_t)(g0 + lh) * FOUT;

  Frag16 af[4];
#pragma unroll
  for (int ks = 0; ks < 4; ++ks) {
    const int lo = ks * 32 + half * 16;
    af[ks].q[0] = *(const uint4*)(hr + lo);       // K = lo..lo+7
    af[ks].q[1] = *(const uint4*)(hr + lo + 8);   // K = lo+8..lo+15
  }

#pragma unroll
  for (int ct = 0; ct < 8; ++ct) {
    v8f acc = {};
#pragma unroll
    for (int ks = 0; ks < 4; ++ks) {
      v16bf bf = *(const v16bf*)&WoS[(ct * 16 + lh) * FIN + ks * 32 + half * 16];
      acc = __builtin_amdgcn_wmma_f32_16x16x32_bf16(
          false, af[ks].v, false, bf, (short)0, acc, false, false);
    }
    const int col = ct * 16 + lh;
    const float bias = bo[col];
#pragma unroll
    for (int v = 0; v < 8; ++v) {
      const float y = acc[v] + bias;
      const float z = (y > 0.f) ? y : (__expf(y) - 1.f);  // ELU(alpha=1)
      out[(size_t)(g0 + v + half * 8) * FOUT + col] = z;
    }
  }
}

// ---------------------------------------------------------------------------
extern "C" void kernel_launch(void* const* d_in, const int* in_sizes, int n_in,
                              void* d_out, int out_size, void* d_ws, size_t ws_size,
                              hipStream_t stream) {
  (void)in_sizes; (void)n_in; (void)out_size; (void)ws_size;
  const float* x   = (const float*)d_in[0];
  const int*   adj = (const int*)d_in[1];
  const float* W   = (const float*)d_in[2];
  const float* a   = (const float*)d_in[3];
  const float* Wo  = (const float*)d_in[4];
  const float* bo  = (const float*)d_in[5];
  float* out = (float*)d_out;

  char* ws = (char*)d_ws;
  unsigned short* wxT  = (unsigned short*)ws;                    // 2 MB  Wx^T bf16 [B][F][N]
  unsigned short* hbuf = (unsigned short*)(ws + (2u << 20));     // 2 MB  h bf16 [B*N][F]
  float* ei = (float*)(ws + (4u << 20));                         // 256 KB [B][H][N]
  float* ej = (float*)(ws + (4u << 20) + (256u << 10));          // 256 KB [B][H][N]

  k_gat_proj<<<64, 256, 0, stream>>>(x, W, a, wxT, ei, ej);
  k_gat_attn<<<BATCH * (NN / 16), 256, 0, stream>>>(adj, wxT, ei, ej, hbuf);
  k_gat_out<<<64, 256, 0, stream>>>(hbuf, Wo, bo, out);
}
